// Head_75015898792097
// MI455X (gfx1250) — matrix operand order repaired and checked by reading, not compiled, not run on hardware
//
#include <hip/hip_runtime.h>

typedef __attribute__((ext_vector_type(16))) _Float16 v16h;
typedef __attribute__((ext_vector_type(8)))  float    v8f;
typedef _Float16 half_t;

#define B_   8
#define T_   2048
#define C_   1024
#define H_   64
#define BTH  (B_*T_*H_)   // 1,048,576 halves per q/k/v buffer
#define HC   (H_*C_)      // 65,536 halves per weight

// ---------------------------------------------------------------------------
// Kernel 0: W fp32 -> fp16 (one shot; halves weight traffic in the GEMM loop)
// ---------------------------------------------------------------------------
__global__ __launch_bounds__(256) void w_to_half(
    const float* __restrict__ Wq, const float* __restrict__ Wk,
    const float* __restrict__ Wv,
    half_t* __restrict__ whq, half_t* __restrict__ whk, half_t* __restrict__ whv)
{
    int i = blockIdx.x * 256 + threadIdx.x;
    if (i < HC) {
        whq[i] = (half_t)Wq[i];
        whk[i] = (half_t)Wk[i];
        whv[i] = (half_t)Wv[i];
    }
}

// ---------------------------------------------------------------------------
// Kernel 1: q,k,v projections.  GEMM M=B*T=16384, N=3*64, K=1024.
// Each wave: one 16-row strip, 12 v8f accumulator tiles, K-loop step 32.
// q,k stored row-major (t,h) f16; v stored transposed (h,t) f16 so the
// attention PV B-operand is a contiguous 32B load per lane.
// ---------------------------------------------------------------------------
__global__ __launch_bounds__(256) void proj_qkv(
    const float* __restrict__ x,
    const half_t* __restrict__ whq, const half_t* __restrict__ whk,
    const half_t* __restrict__ whv,
    half_t* __restrict__ qh, half_t* __restrict__ kh, half_t* __restrict__ vt)
{
    const int wave   = threadIdx.x >> 5;
    const int lane   = threadIdx.x & 31;
    const int laneHi = lane >> 4;        // 0 or 1 (half-wave)
    const int lane16 = lane & 15;
    const int row0   = (blockIdx.x * 8 + wave) * 16;   // 16-row strip

    v8f zero = {};
    v8f acc[12];
#pragma unroll
    for (int i = 0; i < 12; ++i) acc[i] = zero;

    const float*  xr = x + (size_t)(row0 + lane16) * C_;
    const half_t* wptr[3] = { whq, whk, whv };

    for (int k0 = 0; k0 < C_; k0 += 32) {
        // ---- A tile (16x32 f16), lane map: lanes 0-15 K={0..15},
        // lanes 16-31 K={16..31}; convert fp32->f16 in registers.
        const int kb = k0 + laneHi * 16;
        const float4* x4 = (const float4*)(xr + kb);
        float4 A0 = x4[0], A1 = x4[1], A2 = x4[2], A3 = x4[3]; // +0,+4,+8,+12
        v16h a;
        a[0]=(half_t)A0.x; a[1]=(half_t)A0.y; a[2]=(half_t)A0.z; a[3]=(half_t)A0.w;
        a[4]=(half_t)A1.x; a[5]=(half_t)A1.y; a[6]=(half_t)A1.z; a[7]=(half_t)A1.w;
        a[8]=(half_t)A2.x; a[9]=(half_t)A2.y; a[10]=(half_t)A2.z; a[11]=(half_t)A2.w;
        a[12]=(half_t)A3.x; a[13]=(half_t)A3.y; a[14]=(half_t)A3.z; a[15]=(half_t)A3.w;

        // ---- B tiles (32x16 f16): lane holds 16 consecutive K for N=lane%16
        const int koff = k0 + laneHi * 16;
#pragma unroll
        for (int w = 0; w < 3; ++w) {
            const half_t* wb = wptr[w] + (size_t)lane16 * C_ + koff;
#pragma unroll
            for (int n = 0; n < 4; ++n) {
                v16h bt = *(const v16h*)(wb + (size_t)n * 16 * C_);
                acc[w*4+n] = __builtin_amdgcn_wmma_f32_16x16x32_f16(
                    false, a, false, bt, (short)0, acc[w*4+n], false, false);
            }
        }
    }

    // ---- store (C/D layout: VGPR j -> M = j + 8*laneHi, N = lane16)
#pragma unroll
    for (int n = 0; n < 4; ++n) {
        const int col = n * 16 + lane16;
#pragma unroll
        for (int j = 0; j < 8; ++j) {
            const int row = row0 + j + laneHi * 8;
            qh[(size_t)row * H_ + col] = (half_t)acc[0*4+n][j];
            kh[(size_t)row * H_ + col] = (half_t)acc[1*4+n][j];
            const int b = row >> 11;            // T_ = 2048
            const int t = row & (T_ - 1);
            vt[((size_t)b * H_ + col) * T_ + t] = (half_t)acc[2*4+n][j];
        }
    }
}

// ---------------------------------------------------------------------------
// Kernel 2: flash attention, one 16-query tile per wave, 32-key chunks.
// Scores: 4x wmma (K=H=64 chained).  Online softmax via 16-lane shuffles.
// P is turned D-layout -> A-layout through a per-wave LDS tile.
// PV: 4x wmma per chunk into fp32 accumulators; final 1/l normalize.
// ---------------------------------------------------------------------------
__global__ __launch_bounds__(256) void attn_head(
    const half_t* __restrict__ qh, const half_t* __restrict__ kh,
    const half_t* __restrict__ vt, float* __restrict__ out)
{
    __shared__ half_t plds[8][16][32];   // per-wave 16x32 P staging (8 KB)

    const int wave   = threadIdx.x >> 5;
    const int lane   = threadIdx.x & 31;
    const int laneHi = lane >> 4;
    const int lane16 = lane & 15;
    const int gw = blockIdx.x * 8 + wave;     // 0..1023
    const int b  = gw >> 7;                   // /128 query tiles per batch
    const int q0 = (gw & 127) * 16;

    // ---- Q tile in A-layout (two 16x32 halves covering K=h 0..63)
    const half_t* qrow = qh + ((size_t)b * T_ + q0 + lane16) * H_;
    const int kb = laneHi * 16;
    v16h qa0, qa1;
#pragma unroll
    for (int i = 0; i < 8; ++i) {
        qa0[i]     = qrow[kb + i];
        qa0[8 + i] = qrow[kb + 8 + i];
        qa1[i]     = qrow[32 + kb + i];
        qa1[8 + i] = qrow[32 + kb + 8 + i];
    }

    v8f o0 = {}, o1 = {}, o2 = {}, o3 = {};
    float rmax[8], rsum[8];
#pragma unroll
    for (int j = 0; j < 8; ++j) { rmax[j] = -1e30f; rsum[j] = 0.0f; }

    const float scale = 0.125f;               // 1/sqrt(H)
    const half_t* kbatch = kh + (size_t)b * T_ * H_;
    const half_t* vbatch = vt + (size_t)b * H_ * T_;

    for (int s0 = 0; s0 <= q0 + 15; s0 += 32) {
        // ---- scores S = Q K^T for keys [s0, s0+32): B-operand from k rows,
        // h contiguous -> one 32B load per lane per tile.
        const half_t* kr0 = kbatch + (size_t)(s0 + lane16) * H_ + laneHi * 16;
        const half_t* kr1 = kr0 + (size_t)16 * H_;
        v16h kb00 = *(const v16h*)(kr0);        // keys s0..+15, h 0..31
        v16h kb01 = *(const v16h*)(kr0 + 32);   // keys s0..+15, h 32..63
        v16h kb10 = *(const v16h*)(kr1);        // keys +16..+31, h 0..31
        v16h kb11 = *(const v16h*)(kr1 + 32);

        v8f s0t = {}, s1t = {};
        s0t = __builtin_amdgcn_wmma_f32_16x16x32_f16(false, qa0, false, kb00, (short)0, s0t, false, false);
        s0t = __builtin_amdgcn_wmma_f32_16x16x32_f16(false, qa1, false, kb01, (short)0, s0t, false, false);
        s1t = __builtin_amdgcn_wmma_f32_16x16x32_f16(false, qa0, false, kb10, (short)0, s1t, false, false);
        s1t = __builtin_amdgcn_wmma_f32_16x16x32_f16(false, qa1, false, kb11, (short)0, s1t, false, false);

        // ---- scale + causal mask + online softmax (C-layout rows)
        const bool needMask = (s0 + 31 > q0);
        const int key0 = s0 + lane16;
        const int key1 = key0 + 16;
        float p0[8], p1[8], resc[8];
#pragma unroll
        for (int j = 0; j < 8; ++j) {
            const int m = q0 + j + laneHi * 8;
            float va = s0t[j] * scale;
            float vc = s1t[j] * scale;
            if (needMask) {
                if (key0 > m) va = -1e30f;
                if (key1 > m) vc = -1e30f;
            }
            float tm = fmaxf(va, vc);
#pragma unroll
            for (int off = 1; off < 16; off <<= 1)
                tm = fmaxf(tm, __shfl_xor(tm, off, 16));
            const float nm = fmaxf(rmax[j], tm);
            resc[j] = __expf(rmax[j] - nm);
            rmax[j] = nm;
            p0[j] = __expf(va - nm);
            p1[j] = __expf(vc - nm);
            float ps = p0[j] + p1[j];
#pragma unroll
            for (int off = 1; off < 16; off <<= 1)
                ps += __shfl_xor(ps, off, 16);
            rsum[j] = rsum[j] * resc[j] + ps;
        }
#pragma unroll
        for (int j = 0; j < 8; ++j) {
            o0[j] *= resc[j]; o1[j] *= resc[j];
            o2[j] *= resc[j]; o3[j] *= resc[j];
        }

        // ---- P: D-layout -> A-layout via per-wave LDS tile
#pragma unroll
        for (int j = 0; j < 8; ++j) {
            const int m = j + laneHi * 8;
            plds[wave][m][lane16]      = (half_t)p0[j];
            plds[wave][m][lane16 + 16] = (half_t)p1[j];
        }
        __builtin_amdgcn_wave_barrier();     // keep DS store->load in order
        v16h pa;
        const half_t* pr = &plds[wave][lane16][0];
#pragma unroll
        for (int i = 0; i < 8; ++i) {
            pa[i]     = pr[kb + i];
            pa[8 + i] = pr[kb + 8 + i];
        }
        __builtin_amdgcn_wave_barrier();

        // ---- O += P (16x32) x V (32x64): v transposed -> contiguous keys
        const half_t* vr = vbatch + (size_t)lane16 * T_ + s0 + laneHi * 16;
        v16h vb0 = *(const v16h*)(vr);
        v16h vb1 = *(const v16h*)(vr + (size_t)16 * T_);
        v16h vb2 = *(const v16h*)(vr + (size_t)32 * T_);
        v16h vb3 = *(const v16h*)(vr + (size_t)48 * T_);
        o0 = __builtin_amdgcn_wmma_f32_16x16x32_f16(false, pa, false, vb0, (short)0, o0, false, false);
        o1 = __builtin_amdgcn_wmma_f32_16x16x32_f16(false, pa, false, vb1, (short)0, o1, false, false);
        o2 = __builtin_amdgcn_wmma_f32_16x16x32_f16(false, pa, false, vb2, (short)0, o2, false, false);
        o3 = __builtin_amdgcn_wmma_f32_16x16x32_f16(false, pa, false, vb3, (short)0, o3, false, false);
    }

    // ---- normalize and store fp32 output (B,T,H)
#pragma unroll
    for (int j = 0; j < 8; ++j) {
        const float inv = 1.0f / rsum[j];
        const int m = q0 + j + laneHi * 8;
        float* orow = out + ((size_t)b * T_ + m) * H_;
        orow[ 0 + lane16] = o0[j] * inv;
        orow[16 + lane16] = o1[j] * inv;
        orow[32 + lane16] = o2[j] * inv;
        orow[48 + lane16] = o3[j] * inv;
    }
}

// ---------------------------------------------------------------------------
extern "C" void kernel_launch(void* const* d_in, const int* in_sizes, int n_in,
                              void* d_out, int out_size, void* d_ws, size_t ws_size,
                              hipStream_t stream) {
    const float* x  = (const float*)d_in[0];
    const float* Wk = (const float*)d_in[1];
    const float* Wq = (const float*)d_in[2];
    const float* Wv = (const float*)d_in[3];
    float* out = (float*)d_out;

    half_t* ws  = (half_t*)d_ws;
    half_t* qh  = ws;                          // [B*T][H] f16
    half_t* kh  = ws + (size_t)BTH;            // [B*T][H] f16
    half_t* vt  = ws + (size_t)2 * BTH;        // [B][H][T] f16 (transposed)
    half_t* whq = ws + (size_t)3 * BTH;        // [H][C] f16
    half_t* whk = whq + HC;
    half_t* whv = whk + HC;

    w_to_half<<<(HC + 255) / 256, 256, 0, stream>>>(Wq, Wk, Wv, whq, whk, whv);
    proj_qkv<<<128, 256, 0, stream>>>(x, whq, whk, whv, qh, kh, vt);
    attn_head<<<128, 256, 0, stream>>>(qh, kh, vt, out);
}
